// OriginalWindowAttention_82609400971401
// MI455X (gfx1250) — matrix operand order repaired and checked by reading, not compiled, not run on hardware
//
#include <hip/hip_runtime.h>
#include <hip/hip_bf16.h>

typedef __attribute__((ext_vector_type(16))) _Float16 v16h;
typedef __attribute__((ext_vector_type(8)))  _Float16 v8h;
typedef __attribute__((ext_vector_type(4)))  _Float16 v4h;
typedef __attribute__((ext_vector_type(8)))  float    v8f;
typedef __attribute__((ext_vector_type(4)))  float    v4f;

#define WIN_N   49
#define NP      64      // padded token count
#define CDIM    128
#define HEADS   4
#define HD      32
#define LSTR    136     // 128 + 8 pad (f16), 16B-aligned rows, spreads banks
#define VSTR    72      // 64 + 8 pad

// ---------------- fragment load helpers (ISA 7.12.2 16-bit layouts) ----------
// A (16x32, MxK): lane l<16 holds row M=l, K = {0..7, 16..23}; lane l>=16 holds
// row M=l-16, K = {8..15, 24..31}.
__device__ __forceinline__ v16h lda16(const _Float16* base, int stride) {
  const int l  = threadIdx.x & 31;
  const int m  = l & 15;
  const int hf = l >> 4;
  const _Float16* p = base + m * stride + hf * 8;
  v8h lo = *(const v8h*)(p);
  v8h hi = *(const v8h*)(p + 16);
  return __builtin_shufflevector(lo, hi, 0,1,2,3,4,5,6,7,8,9,10,11,12,13,14,15);
}

// B (32x16, KxN): lane l<16 holds column N=l, K = {0..7, 16..23}; lane l>=16 holds
// column N=l-16, K = {8..15, 24..31}. Source stored n-major ("B^T").
__device__ __forceinline__ v16h ldb16(const _Float16* base, int stride) {
  const int l  = threadIdx.x & 31;
  const int n  = l & 15;
  const int hf = l >> 4;
  const _Float16* p = base + n * stride + hf * 8;
  v8h lo = *(const v8h*)(p);
  v8h hi = *(const v8h*)(p + 16);
  return __builtin_shufflevector(lo, hi, 0,1,2,3,4,5,6,7,8,9,10,11,12,13,14,15);
}

#define WMMA_F16(a, b, c) \
  __builtin_amdgcn_wmma_f32_16x16x32_f16(false, (a), false, (b), (short)0, (c), false, false)

// ---------------- prep kernel ------------------------------------------------
// Converts weights to f16 and pre-swizzles bias/mask into D-fragment order:
//   frag[tile(qt,kt)][lane][j]  with row = qt*16 + (lane>>4)*8 + j,
//                                   col = kt*16 + (lane&15)
// so the main kernel adds them with aligned v8f loads (no gathers, no guards).
__global__ void prep_kernel(const float* __restrict__ Wqkv,
                            const float* __restrict__ Wproj,
                            const float* __restrict__ rel_table,
                            const float* __restrict__ mask,
                            _Float16* __restrict__ wqkv_h,
                            _Float16* __restrict__ wproj_h,
                            float* __restrict__ biasF,
                            float* __restrict__ maskF,
                            int num_win) {
  const int t  = blockIdx.x * blockDim.x + threadIdx.x;
  const int gs = gridDim.x * blockDim.x;
  for (int i = t; i < 3 * CDIM * CDIM; i += gs) wqkv_h[i]  = (_Float16)Wqkv[i];
  for (int i = t; i < CDIM * CDIM;     i += gs) wproj_h[i] = (_Float16)Wproj[i];

  // biasF[h][qt][kt][lane*8+j] : rel-pos bias, padded cols forced to -1e30
  for (int i = t; i < HEADS * 4 * 4 * 256; i += gs) {
    const int j    = i & 7;
    const int lane = (i >> 3) & 31;
    const int kt   = (i >> 8) & 3;
    const int qt   = (i >> 10) & 3;
    const int h    = i >> 12;
    const int row  = qt * 16 + (lane >> 4) * 8 + j;
    const int col  = kt * 16 + (lane & 15);
    float v;
    if (col >= WIN_N)      v = -1e30f;   // kill padded key columns in softmax
    else if (row >= WIN_N) v = 0.0f;     // padded query rows: keep finite
    else {
      const int ih = row / 7, iw = row % 7;
      const int jh = col / 7, jw = col % 7;
      const int rel = (ih - jh + 6) * 13 + (iw - jw + 6);
      v = rel_table[rel * HEADS + h];
    }
    biasF[i] = v;
  }

  // maskF[w][qt][kt][lane*8+j] : window mask, zero in padding
  for (int i = t; i < num_win * 4 * 4 * 256; i += gs) {
    const int j    = i & 7;
    const int lane = (i >> 3) & 31;
    const int kt   = (i >> 8) & 3;
    const int qt   = (i >> 10) & 3;
    const int w    = i >> 12;
    const int row  = qt * 16 + (lane >> 4) * 8 + j;
    const int col  = kt * 16 + (lane & 15);
    maskF[i] = (row < WIN_N && col < WIN_N)
                 ? mask[(size_t)w * WIN_N * WIN_N + row * WIN_N + col] : 0.0f;
  }
}

// ---------------- fused window-attention kernel ------------------------------
__launch_bounds__(256)
__global__ void winattn_kernel(const float* __restrict__ x,
                               const float* __restrict__ bqkv,
                               const float* __restrict__ bproj,
                               const _Float16* __restrict__ wqkv_h,
                               const _Float16* __restrict__ wproj_h,
                               const float* __restrict__ biasF,
                               const float* __restrict__ maskF,
                               float* __restrict__ out,
                               int num_win) {
  // 89 KB LDS; xh region is reused as oh after phase 1.
  __shared__ __align__(16) _Float16 smem[44544];
  _Float16* xh = smem;                 // 64*136  (phase 0/1; reused as oh)
  _Float16* qh = smem + 8704;          // 64*136  (pre-scaled by 1/sqrt(hd))
  _Float16* kh = smem + 17408;         // 64*136
  _Float16* vT = smem + 26112;         // 128*72  (dim-major: B^T for attn@V)
  _Float16* aw = smem + 35328;         // 8*16*72 (per-wave attn relayout scratch)
  _Float16* oh = xh;

  const int b   = blockIdx.x;
  const int tid = threadIdx.x;
  // Force wave-uniform values into SGPRs so loops/branches go scalar.
  const int wv  = __builtin_amdgcn_readfirstlane(tid >> 5);   // wave id 0..7
  const int l   = tid & 31;
  const int ln  = l & 15;              // D-tile column within tile
  const int mh  = (l >> 4) * 8;        // D-tile row offset of this half-wave

  // ---- phase 0: load x (49x128 f32) -> f16 LDS, zero-pad rows 49..63 -------
  // x is streamed exactly once -> non-temporal loads keep weights hot in L2.
  const float* xb = x + (size_t)b * WIN_N * CDIM;
  for (int i = tid; i < NP * 32; i += 256) {     // 32 float4 per row
    const int r  = i >> 5;
    const int c4 = (i & 31) << 2;
    v4f v = {};
    if (r < WIN_N) v = __builtin_nontemporal_load((const v4f*)(xb + r * CDIM + c4));
    v4h p;
#pragma unroll
    for (int k = 0; k < 4; ++k) p[k] = (_Float16)v[k];
    *(v4h*)(xh + r * LSTR + c4) = p;             // packed 8B LDS store
  }
  __syncthreads();

  // ---- phase 1: qkv = x @ WqkvT + bqkv  (4x24 tiles, K=128) ----------------
  // mt = t&3 is loop-invariant (stride 8) -> A fragments stay in registers.
  // B fragments preloaded as a batch -> one 8x b128 load clause per tile.
  const float qscale = 0.17677669529663689f;  // 1/sqrt(32)
  for (int t = wv; t < 96; t += 8) {
    const int mt = t & 3;
    const int nt = t >> 2;
    v16h bbv[4];
#pragma unroll
    for (int kk = 0; kk < 4; ++kk)
      bbv[kk] = ldb16(wqkv_h + (nt * 16) * CDIM + kk * 32, CDIM);
    v8f acc = {};
#pragma unroll
    for (int kk = 0; kk < 4; ++kk) {
      v16h a = lda16(xh + (mt * 16) * LSTR + kk * 32, LSTR);
      acc = WMMA_F16(a, bbv[kk], acc);
    }
    const float bi  = bqkv[nt * 16 + ln];
    const int sec   = nt >> 3;              // uniform: 0=q 1=k 2=v
    const int cc    = (nt & 7) * 16 + ln;   // column within section
    const int row0  = mt * 16 + mh;
    if (sec == 0) {
#pragma unroll
      for (int j = 0; j < 8; ++j)
        qh[(row0 + j) * LSTR + cc] = (_Float16)((acc[j] + bi) * qscale);
    } else if (sec == 1) {
#pragma unroll
      for (int j = 0; j < 8; ++j)
        kh[(row0 + j) * LSTR + cc] = (_Float16)(acc[j] + bi);
    } else {
      // transposed store: 8 contiguous tokens -> one packed 16B LDS store
      v8h pk;
#pragma unroll
      for (int j = 0; j < 8; ++j) pk[j] = (_Float16)(acc[j] + bi);
      *(v8h*)(vT + cc * VSTR + row0) = pk;
    }
  }
  __syncthreads();

  // ---- phase 2: scores + softmax (in registers) + attn @ V -----------------
  const int w = b % num_win;
  for (int t = wv; t < 16; t += 8) {                 // (head, q-tile) combos
    const int h = t >> 2, qt = t & 3;
    const v16h aq = lda16(qh + (qt * 16) * LSTR + h * HD, LSTR);
    v8f s[4];
#pragma unroll
    for (int kt = 0; kt < 4; ++kt) {                 // K = hd = 32 -> one WMMA
      v16h bk = ldb16(kh + (kt * 16) * LSTR + h * HD, LSTR);
      v8f z = {};
      s[kt] = WMMA_F16(aq, bk, z);
    }
    // dense fragment-ordered bias + mask adds: 2 aligned v8f loads per k-tile
    const float* bf = biasF + (((h * 4 + qt) * 4) << 8) + l * 8;
    const float* mf = maskF + (((w * 4 + qt) * 4) << 8) + l * 8;
#pragma unroll
    for (int kt = 0; kt < 4; ++kt) {
      const v8f bv = *(const v8f*)(bf + (kt << 8));
      const v8f mv = *(const v8f*)(mf + (kt << 8));
#pragma unroll
      for (int j = 0; j < 8; ++j) s[kt][j] += bv[j] + mv[j];
    }
    // softmax over 64 key columns; a row lives in one 16-lane half, so xor
    // shuffles with masks 8..1 reduce it without crossing halves.
    float rs[8];
#pragma unroll
    for (int j = 0; j < 8; ++j) {
      float m = fmaxf(fmaxf(s[0][j], s[1][j]), fmaxf(s[2][j], s[3][j]));
#pragma unroll
      for (int off = 8; off >= 1; off >>= 1) m = fmaxf(m, __shfl_xor(m, off, 32));
      float sum = 0.0f;
#pragma unroll
      for (int kt = 0; kt < 4; ++kt) {
        const float e = __expf(s[kt][j] - m);
        s[kt][j] = e;
        sum += e;
      }
#pragma unroll
      for (int off = 8; off >= 1; off >>= 1) sum += __shfl_xor(sum, off, 32);
      rs[j] = __builtin_amdgcn_rcpf(sum);
    }
    // relayout attn D-tile -> A-fragment via wave-private LDS (same-wave LDS
    // ops are in order; no workgroup barrier needed)
    _Float16* a_w = aw + wv * 16 * VSTR;
#pragma unroll
    for (int kt = 0; kt < 4; ++kt)
#pragma unroll
      for (int j = 0; j < 8; ++j)
        a_w[(mh + j) * VSTR + kt * 16 + ln] = (_Float16)(s[kt][j] * rs[j]);

    // out_tile = attn(16x64) @ v(64x32): 2 N-tiles x 2 K-chunks
#pragma unroll
    for (int ntl = 0; ntl < 2; ++ntl) {
      v8f o = {};
#pragma unroll
      for (int kc = 0; kc < 2; ++kc) {
        v16h aa = lda16(a_w + kc * 32, VSTR);
        v16h bv = ldb16(vT + (h * HD + ntl * 16) * VSTR + kc * 32, VSTR);
        o = WMMA_F16(aa, bv, o);
      }
#pragma unroll
      for (int j = 0; j < 8; ++j)
        oh[(qt * 16 + mh + j) * LSTR + h * HD + ntl * 16 + ln] = (_Float16)o[j];
    }
  }
  __syncthreads();

  // ---- phase 3: out = oh @ WprojT + bproj  (4x8 tiles, K=128) --------------
  float* ob = out + (size_t)b * WIN_N * CDIM;
  for (int t = wv; t < 32; t += 8) {
    const int mt = t & 3, nt = t >> 2;
    v16h bbv[4];
#pragma unroll
    for (int kk = 0; kk < 4; ++kk)
      bbv[kk] = ldb16(wproj_h + (nt * 16) * CDIM + kk * 32, CDIM);
    v8f acc = {};
#pragma unroll
    for (int kk = 0; kk < 4; ++kk) {
      v16h a = lda16(oh + (mt * 16) * LSTR + kk * 32, LSTR);
      acc = WMMA_F16(a, bbv[kk], acc);
    }
    const float bi = bproj[nt * 16 + ln];
#pragma unroll
    for (int j = 0; j < 8; ++j) {
      const int row = mt * 16 + mh + j;
      if (row < WIN_N)
        __builtin_nontemporal_store(acc[j] + bi, &ob[row * CDIM + nt * 16 + ln]);
    }
  }
}

// ---------------- host launch -----------------------------------------------
extern "C" void kernel_launch(void* const* d_in, const int* in_sizes, int n_in,
                              void* d_out, int out_size, void* d_ws, size_t ws_size,
                              hipStream_t stream) {
  const float* x         = (const float*)d_in[0];
  const float* mask      = (const float*)d_in[1];
  const float* Wqkv      = (const float*)d_in[2];
  const float* bqkv      = (const float*)d_in[3];
  const float* Wproj     = (const float*)d_in[4];
  const float* bproj     = (const float*)d_in[5];
  const float* rel_table = (const float*)d_in[6];
  float* out = (float*)d_out;

  const int B       = in_sizes[0] / (WIN_N * CDIM);          // 4096
  const int num_win = in_sizes[1] / (WIN_N * WIN_N);         // 64

  // workspace: f16 weights + fragment-ordered bias/mask tables (~1.2 MB)
  char* ws = (char*)d_ws;
  _Float16* wqkv_h  = (_Float16*)ws;                         // 98304 B
  _Float16* wproj_h = (_Float16*)(ws + 98304);               // 32768 B
  float*    biasF   = (float*)(ws + 131072);                 // 65536 B
  float*    maskF   = (float*)(ws + 196608);                 // num_win*16384 B

  prep_kernel<<<128, 256, 0, stream>>>(Wqkv, Wproj, rel_table, mask,
                                       wqkv_h, wproj_h, biasF, maskF, num_win);
  winattn_kernel<<<B, 256, 0, stream>>>(x, bqkv, bproj, wqkv_h, wproj_h,
                                        biasF, maskF, out, num_win);
}
